// StyleWSConv_27169963115032
// MI455X (gfx1250) — matrix operand order repaired and checked by reading, not compiled, not run on hardware
//
#include <hip/hip_runtime.h>
#include <hip/hip_bf16.h>

// ---------------------------------------------------------------------------
// StyleGAN2-style modulated 3x3 conv as implicit GEMM on CDNA5 WMMA (gfx1250).
//   b=8, cin=cout=512, H=W=64, k=3  ->  155 GFLOP, compute bound on MI455X.
// v_wmma_f32_16x16x32_f16 with f32 accumulation; f16 modulated weights.
// Per block: 128 cout x 128 pixels (two image rows). 4 input dy-rows x 32
// channels staged per K-step into double-buffered LDS; all 9 taps x 2 output
// rows consume shifted views (A-fragment reused by 8 WMMAs, 1 barrier/K-step).
// ---------------------------------------------------------------------------

typedef _Float16 v16h __attribute__((ext_vector_type(16)));
typedef _Float16 v8h  __attribute__((ext_vector_type(8)));
typedef _Float16 v2h  __attribute__((ext_vector_type(2)));
typedef float    v8f  __attribute__((ext_vector_type(8)));

#define CIN   512
#define COUT  512
#define NB    8
#define HW    4096          // 64*64
#define KTAPS 9
#define KTOT  (CIN*KTAPS)   // 4608
#define COLS  66            // 64 pixels + 1 zero col each side
#define CSTR  40            // halves per column: 32 data + 8 pad (80 B, 16B-aligned)
#define RSTR  (COLS*CSTR)   // 2640 halves per dy-row (5280 B, 16B-aligned)
#define NROWS 4             // input rows staged (2 output rows + halo)
#define BUFH  (NROWS*RSTR)  // 10560 halves per buffer (21120 B, 16B-aligned)

// ---------------------------------------------------------------------------
// Per-sample style inf-norm:  smax[b] = max|style[b,:]|
// ---------------------------------------------------------------------------
__global__ void swsc_style_max(const float* __restrict__ style,
                               float* __restrict__ smax) {
    __shared__ float red[256];
    const int b = blockIdx.x, t = threadIdx.x;
    float m = 0.f;
    for (int i = t; i < CIN; i += 256) m = fmaxf(m, fabsf(style[b * CIN + i]));
    red[t] = m; __syncthreads();
    for (int s = 128; s > 0; s >>= 1) {
        if (t < s) red[t] = fmaxf(red[t], red[t + s]);
        __syncthreads();
    }
    if (t == 0) smax[b] = red[0];
}

// ---------------------------------------------------------------------------
// Per-filter weight inf-norm: wmax[co] = max|weight[co,:,:,:]|  (4608 elems)
// ---------------------------------------------------------------------------
__global__ void swsc_filter_max(const float* __restrict__ weight,
                                float* __restrict__ wmax) {
    __shared__ float red[256];
    const int co = blockIdx.x, t = threadIdx.x;
    float m = 0.f;
#pragma unroll
    for (int j = 0; j < 18; ++j)       // 18*256 == 4608 exactly
        m = fmaxf(m, fabsf(weight[(size_t)co * KTOT + t + j * 256]));
    red[t] = m; __syncthreads();
    for (int s = 128; s > 0; s >>= 1) {
        if (t < s) red[t] = fmaxf(red[t], red[t + s]);
        __syncthreads();
    }
    if (t == 0) wmax[co] = red[0];
}

// ---------------------------------------------------------------------------
// Modulate + demodulate, emit f16 weights laid out [b][cout][tap][cin]
// (cin contiguous -> aligned 16B A-fragment chunks). The extra x*scale of the
// reference is folded in here.
// ---------------------------------------------------------------------------
__global__ void swsc_modweight(const float* __restrict__ weight,
                               const float* __restrict__ style,
                               const float* __restrict__ wmax,
                               const float* __restrict__ smax,
                               _Float16* __restrict__ wmod) {
    const int co = blockIdx.x, b = blockIdx.y, t = threadIdx.x;
    const float scale = 0.014731391f;            // 1/sqrt(cin*k*k) = 1/sqrt(4608)
    const float wsc  = scale / wmax[co];
    const float sinv = 1.0f / smax[b];

    float vals[18];
    float ss = 0.f;
#pragma unroll
    for (int j = 0; j < 18; ++j) {
        const int i   = t + j * 256;             // i in [0,4608)
        const int cin = i / 9;
        const float v = weight[(size_t)co * KTOT + i] * wsc
                      * (style[b * CIN + cin] * sinv);
        vals[j] = v;
        ss += v * v;
    }
    __shared__ float red[256];
    red[t] = ss; __syncthreads();
    for (int s = 128; s > 0; s >>= 1) {
        if (t < s) red[t] += red[t + s];
        __syncthreads();
    }
    const float demod = rsqrtf(red[0] + 1e-8f) * scale;   // fold x-scale
#pragma unroll
    for (int j = 0; j < 18; ++j) {
        const int i = t + j * 256;
        const int cin = i / 9, kk = i - cin * 9;
        wmod[(((size_t)b * COUT + co) * KTAPS + kk) * CIN + cin] =
            (_Float16)(vals[j] * demod);
    }
}

// ---------------------------------------------------------------------------
// Main conv: grid = (cout/128, 32 row-pairs, 8 samples), 256 thr = 8 waves.
// K-step = 32 channels; per K-step stage [4 dy-rows][66 cols][32 ch] f16 into
// LDS (double-buffered), then 9 taps x 2 out-rows x 4 N-subtiles = 72 WMMAs.
// Wave w owns cout rows [mblk*128 + w*16, +16); A fragment shared by 8 WMMAs.
// ---------------------------------------------------------------------------
__global__ __launch_bounds__(256) void
swsc_conv_wmma(const float* __restrict__ x,
               const _Float16* __restrict__ wmod,
               const float* __restrict__ bias,
               const float* __restrict__ noise,
               const float* __restrict__ nstrength,
               float* __restrict__ out) {
    const int mblk = blockIdx.x;          // cout block of 128
    const int y0   = blockIdx.y * 2;      // first of two output rows
    const int b    = blockIdx.z;          // sample
    const int t    = threadIdx.x;
    const int lane = t & 31;              // wave32
    const int wv   = t >> 5;
    const int nIn16 = lane & 15;          // M row (A) / N col (B,C) within tile
    const int grp   = lane >> 4;          // K-half selector

    __shared__ _Float16 ldsX[2 * BUFH];   // 42.2 KB of the 320 KB/WGP

    v8f acc[8] = {v8f{}, v8f{}, v8f{}, v8f{}, v8f{}, v8f{}, v8f{}, v8f{}};

    // This lane's A-matrix row (16-bit A 16x32 layout: lane%16 = M).
    const int mrow = mblk * 128 + wv * 16 + nIn16;
    const _Float16* __restrict__ wrow =
        wmod + ((size_t)b * COUT + mrow) * (size_t)(KTAPS * CIN);
    const float* __restrict__ xb = x + (size_t)b * CIN * HW;

    // Staging decomposition: 4 rows * 16 kpairs * 64 cols = 16*256 slots.
    const int sCol = t & 63;              // image column 0..63 -> LDS col+1
    const int sRK  = t >> 6;              // rk = sRK + it*4 ; kp = rk&15, r = rk>>4

    auto stage = [&](int buf, int c0) {
        _Float16* __restrict__ dst = ldsX + buf * BUFH;
#pragma unroll
        for (int it = 0; it < 16; ++it) {
            const int rk = sRK + it * 4;
            const int kp = rk & 15;       // channel pair 0..15
            const int r  = rk >> 4;       // dy-row 0..3
            const int yy = y0 + r - 1;    // input row
            const int c  = c0 + kp * 2;
            float f0 = 0.f, f1 = 0.f;
            if (yy >= 0 && yy < 64) {
                const float* p = xb + (size_t)c * HW + yy * 64 + sCol;  // coalesced
                f0 = p[0];
                f1 = p[HW];
            }
            v2h pk; pk.x = (_Float16)f0; pk.y = (_Float16)f1;
            *(v2h*)&dst[r * RSTR + (sCol + 1) * CSTR + kp * 2] = pk;
        }
        if (t < 128) {                     // zero border columns (LDS col 0, 65)
            const int colb = (t & 1) ? 65 : 0;
            const int rk = t >> 1;         // 0..63
            const int kp = rk & 15, r = rk >> 4;
            v2h z; z.x = (_Float16)0.f; z.y = (_Float16)0.f;
            *(v2h*)&dst[r * RSTR + colb * CSTR + kp * 2] = z;
        }
    };

    auto compute = [&](int buf, int c0) {
        const _Float16* __restrict__ src = ldsX + buf * BUFH;
#pragma unroll
        for (int tap = 0; tap < KTAPS; ++tap) {
            const int rb = tap / 3;        // dy offset (input row = rb + o)
            const int dx = tap % 3 - 1;
            // A fragment: K = khalf*8 + {0..7} and 16 + khalf*8 + {0..7}
            const _Float16* aptr = wrow + tap * CIN + c0 + grp * 8;
            const v8h alo = *(const v8h*)(aptr);
            const v8h ahi = *(const v8h*)(aptr + 16);
            const v16h afrag = __builtin_shufflevector(
                alo, ahi, 0, 1, 2, 3, 4, 5, 6, 7, 8, 9, 10, 11, 12, 13, 14, 15);
#pragma unroll
            for (int o = 0; o < 2; ++o) {  // output row within block
                const _Float16* __restrict__ rowb = src + (rb + o) * RSTR;
#pragma unroll
                for (int nt = 0; nt < 4; ++nt) {
                    const int colx = nt * 16 + nIn16 + dx + 1;   // 0..65
                    const _Float16* bp = &rowb[colx * CSTR + grp * 8];
                    const v8h blo = *(const v8h*)bp;             // K = kgrp*8 + 0..7
                    const v8h bhi = *(const v8h*)(bp + 16);      // K = 16 + kgrp*8 + 0..7
                    const v16h bfrag = __builtin_shufflevector(
                        blo, bhi, 0, 1, 2, 3, 4, 5, 6, 7,
                        8, 9, 10, 11, 12, 13, 14, 15);
                    acc[o * 4 + nt] = __builtin_amdgcn_wmma_f32_16x16x32_f16(
                        false, afrag, false, bfrag, (short)0, acc[o * 4 + nt],
                        false, false);
                }
            }
        }
    };

    // Double-buffered K loop: 16 chunks of 32 channels, 1 barrier per chunk.
    stage(0, 0);
    for (int i = 0; i < 16; ++i) {
        __syncthreads();                   // staging of buf(i) visible; reads of
                                           // buf(i-1) complete before overwrite
        if (i + 1 < 16) stage((i + 1) & 1, (i + 1) * 32);
        compute(i & 1, i * 32);
    }

    // ---- epilogue: C layout is VGPR r, lane l -> M = r + 8*(l/16), N = l%16
    const float ns = nstrength[0];
#pragma unroll
    for (int o = 0; o < 2; ++o) {
        const int p0 = (y0 + o) * 64;
#pragma unroll
        for (int nt = 0; nt < 4; ++nt) {
            const int col = nt * 16 + nIn16;
            const float nz = noise[(size_t)b * HW + p0 + col] * ns;
#pragma unroll
            for (int r = 0; r < 8; ++r) {
                const int m = mblk * 128 + wv * 16 + grp * 8 + r;
                // streaming store: keep x/wmod resident in the 192 MB L2
                __builtin_nontemporal_store(acc[o * 4 + nt][r] + bias[m] + nz,
                    &out[((size_t)b * COUT + m) * HW + p0 + col]);
            }
        }
    }
}

// ---------------------------------------------------------------------------
// Launch: inputs are (x, style, noise, weight, bias, noise_strength), all f32.
// Workspace: [0,2KB) wmax, [2KB,4KB) smax, [4KB, 4KB+37.75MB) f16 wmod.
// ---------------------------------------------------------------------------
extern "C" void kernel_launch(void* const* d_in, const int* in_sizes, int n_in,
                              void* d_out, int out_size, void* d_ws, size_t ws_size,
                              hipStream_t stream) {
    const float* x      = (const float*)d_in[0];
    const float* style  = (const float*)d_in[1];
    const float* noise  = (const float*)d_in[2];
    const float* weight = (const float*)d_in[3];
    const float* bias   = (const float*)d_in[4];
    const float* nstr   = (const float*)d_in[5];
    float* out = (float*)d_out;

    float*    wmax = (float*)d_ws;                       // 512 floats
    float*    smax = wmax + 512;                         // 8 floats
    _Float16* wmod = (_Float16*)((char*)d_ws + 4096);    // 8*512*9*512 halves

    swsc_style_max <<<NB,   256, 0, stream>>>(style, smax);
    swsc_filter_max<<<COUT, 256, 0, stream>>>(weight, wmax);
    swsc_modweight <<<dim3(COUT, NB), 256, 0, stream>>>(weight, style, wmax, smax, wmod);
    swsc_conv_wmma <<<dim3(COUT / 128, 32, NB), 256, 0, stream>>>(
        x, wmod, bias, noise, nstr, out);
}
